// LSTM_encoder_decoder_7181185319032
// MI455X (gfx1250) — matrix operand order repaired and checked by reading, not compiled, not run on hardware
//
#include <hip/hip_runtime.h>
#include <hip/hip_bf16.h>
#include <stdint.h>

#define Hdim 256
#define Bdim 1024
#define TSRC 128
#define TTGT 64

typedef __attribute__((ext_vector_type(16))) __bf16 v16bf;
typedef __attribute__((ext_vector_type(8)))  float  v8f;
typedef __attribute__((ext_vector_type(4)))  unsigned int u32x4;

union FragU { v16bf v; u32x4 q[2]; };

__device__ __forceinline__ unsigned short f2bf(float f) {
  unsigned int u = __float_as_uint(f);
  u += 0x7FFFu + ((u >> 16) & 1u);          // round-to-nearest-even
  return (unsigned short)(u >> 16);
}
__device__ __forceinline__ float bf2f(unsigned short s) {
  return __uint_as_float(((unsigned int)s) << 16);
}
__device__ __forceinline__ float sigf(float v) { return 1.0f / (1.0f + __expf(-v)); }

// A fragment: 16x32 bf16 tile of row-major [Bdim x Hdim] matrix.
// lane<16: rows m0+lane, K = k0+0..15 (contiguous)
// lane>=16: same rows, K = k0+16..31 (contiguous)
__device__ __forceinline__ v16bf load_a_frag(const unsigned short* __restrict__ base,
                                             int m0, int k0, int lane) {
  int mm = m0 + (lane & 15);
  int koff = k0 + ((lane >> 4) << 4);
  const unsigned short* p = base + mm * Hdim + koff;
  FragU f;
  f.q[0] = *reinterpret_cast<const u32x4*>(p);       // 8 bf16
  f.q[1] = *reinterpret_cast<const u32x4*>(p + 8);   // 8 bf16, K+8
  return f.v;
}

// B fragment: 32x16 bf16 tile of W^T, W row-major [4H x Hdim].
// lane l holds column n = row0+(l&15) of B  == row (row0+(l&15)) of W,
// K = k0 + (l>>4)*16 + {0..15}  -> 16 contiguous bf16 in memory.
__device__ __forceinline__ v16bf load_b_frag(const unsigned short* __restrict__ Wb,
                                             int row0, int k0, int lane) {
  int rr = row0 + (lane & 15);
  int koff = k0 + ((lane >> 4) << 4);
  const unsigned short* p = Wb + rr * Hdim + koff;
  FragU f;
  f.q[0] = *reinterpret_cast<const u32x4*>(p);
  f.q[1] = *reinterpret_cast<const u32x4*>(p + 8);
  return f.v;
}

// Fused LSTM cell: G = hl@Wih^T + hp@Whh^T (+ x@WihSmall^T) + b, then gate math.
// One wave -> one 16x16 (batch x hidden) chunk, all 4 gates in registers.
__global__ __launch_bounds__(128)
void lstm_cell_kernel(const unsigned short* __restrict__ hp,   // h_prev  [Bdim x Hdim] bf16
                      const unsigned short* __restrict__ Whh,  // [4H x Hdim] bf16
                      const unsigned short* __restrict__ hl,   // lower-layer h or null
                      const unsigned short* __restrict__ Wih,  // [4H x Hdim] bf16 or null
                      const float* __restrict__ x,             // raw input or null
                      int x_stride, int x_off,
                      const float* __restrict__ WihS,          // [4H x 3] f32 (with x)
                      const float* __restrict__ bias,          // [4H]
                      float* __restrict__ c,                   // in/out [Bdim x Hdim] f32
                      unsigned short* __restrict__ h_out)      // [Bdim x Hdim] bf16
{
  int wave = (blockIdx.x * blockDim.x + threadIdx.x) >> 5;
  int lane = threadIdx.x & 31;
  int m0 = (wave >> 4) << 4;   // 64 batch tiles
  int n0 = (wave & 15) << 4;   // 16 hidden tiles

  v8f acc[4] = {v8f{0}, v8f{0}, v8f{0}, v8f{0}};  // i, f, g, o

  // recurrent GEMM: hp @ Whh^T
  for (int k0 = 0; k0 < Hdim; k0 += 32) {
    v16bf a = load_a_frag(hp, m0, k0, lane);
#pragma unroll
    for (int g = 0; g < 4; ++g) {
      v16bf b = load_b_frag(Whh, g * Hdim + n0, k0, lane);
      acc[g] = __builtin_amdgcn_wmma_f32_16x16x32_bf16(false, a, false, b,
                                                       (short)0, acc[g], false, false);
    }
  }
  // layer-1 input GEMM: hl @ Wih^T (uniform branch; EXEC stays all-ones)
  if (hl != nullptr) {
    for (int k0 = 0; k0 < Hdim; k0 += 32) {
      v16bf a = load_a_frag(hl, m0, k0, lane);
#pragma unroll
      for (int g = 0; g < 4; ++g) {
        v16bf b = load_b_frag(Wih, g * Hdim + n0, k0, lane);
        acc[g] = __builtin_amdgcn_wmma_f32_16x16x32_bf16(false, a, false, b,
                                                         (short)0, acc[g], false, false);
      }
    }
  }

  // epilogue: C/D layout -> element (m0 + r + (lane>>4)*8, n0 + (lane&15))
  int nc = n0 + (lane & 15);
  int mbase = m0 + ((lane >> 4) << 3);
  float bI = bias[0 * Hdim + nc];
  float bF = bias[1 * Hdim + nc];
  float bG = bias[2 * Hdim + nc];
  float bO = bias[3 * Hdim + nc];
  float wI[3], wF[3], wG[3], wO[3];
  if (x != nullptr) {
#pragma unroll
    for (int k = 0; k < 3; ++k) {
      wI[k] = WihS[(0 * Hdim + nc) * 3 + k];
      wF[k] = WihS[(1 * Hdim + nc) * 3 + k];
      wG[k] = WihS[(2 * Hdim + nc) * 3 + k];
      wO[k] = WihS[(3 * Hdim + nc) * 3 + k];
    }
  }
#pragma unroll
  for (int r = 0; r < 8; ++r) {
    int m = mbase + r;
    float gi = acc[0][r] + bI;
    float gf = acc[1][r] + bF;
    float gg = acc[2][r] + bG;
    float go = acc[3][r] + bO;
    if (x != nullptr) {
      float x0 = x[m * x_stride + x_off + 0];
      float x1 = x[m * x_stride + x_off + 1];
      float x2 = x[m * x_stride + x_off + 2];
      gi += x0 * wI[0] + x1 * wI[1] + x2 * wI[2];
      gf += x0 * wF[0] + x1 * wF[1] + x2 * wF[2];
      gg += x0 * wG[0] + x1 * wG[1] + x2 * wG[2];
      go += x0 * wO[0] + x1 * wO[1] + x2 * wO[2];
    }
    float co = c[m * Hdim + nc];
    float cn = sigf(gf) * co + sigf(gi) * tanhf(gg);
    float hn = sigf(go) * tanhf(cn);
    c[m * Hdim + nc] = cn;
    h_out[m * Hdim + nc] = f2bf(hn);
  }
}

// Decoder head: out = h1 @ linW^T + b; build next dec_in. One wave per batch row.
__global__ __launch_bounds__(128)
void dec_out_kernel(const unsigned short* __restrict__ h1,
                    const float* __restrict__ linW, const float* __restrict__ linb,
                    float* __restrict__ dec_in, float* __restrict__ out, int t) {
  int wave = (blockIdx.x * blockDim.x + threadIdx.x) >> 5;
  int lane = threadIdx.x & 31;
  int m = wave;
  float s = 0.0f;
  for (int k = lane; k < Hdim; k += 32) s += bf2f(h1[m * Hdim + k]) * linW[k];
#pragma unroll
  for (int off = 16; off > 0; off >>= 1) s += __shfl_down(s, off, 32);
  if (lane == 0) {
    float o0 = s + linb[0];
    float x0 = dec_in[m * 3 + 0];
    float x1 = dec_in[m * 3 + 1];
    float s1 = x0 - o0;
    float s2 = x1 - s1;
    dec_in[m * 3 + 0] = o0;
    dec_in[m * 3 + 1] = s1;
    dec_in[m * 3 + 2] = s2;
    out[m * TTGT + t] = o0;
  }
}

__global__ void f2bf_kernel(const float* __restrict__ in, unsigned short* __restrict__ out, int n) {
  int i = blockIdx.x * blockDim.x + threadIdx.x;
  if (i < n) out[i] = f2bf(in[i]);
}
__global__ void zero_f32_kernel(float* p, int n) {
  int i = blockIdx.x * blockDim.x + threadIdx.x;
  if (i < n) p[i] = 0.0f;
}
__global__ void zero_u16_kernel(unsigned short* p, int n) {
  int i = blockIdx.x * blockDim.x + threadIdx.x;
  if (i < n) p[i] = 0;
}
__global__ void decin_init_kernel(const float* __restrict__ x, float* __restrict__ dec_in) {
  int i = blockIdx.x * blockDim.x + threadIdx.x;
  if (i < Bdim * 3) {
    int m = i / 3, k = i % 3;
    dec_in[i] = x[(m * TSRC + (TSRC - 1)) * 3 + k];  // x[:, -1, :]
  }
}

extern "C" void kernel_launch(void* const* d_in, const int* in_sizes, int n_in,
                              void* d_out, int out_size, void* d_ws, size_t ws_size,
                              hipStream_t stream) {
  const float* x        = (const float*)d_in[0];
  const float* enc_Wih0 = (const float*)d_in[2];
  const float* enc_Wih1 = (const float*)d_in[3];
  const float* enc_Whh  = (const float*)d_in[4];
  const float* enc_b    = (const float*)d_in[5];
  const float* dec_Wih0 = (const float*)d_in[6];
  const float* dec_Wih1 = (const float*)d_in[7];
  const float* dec_Whh  = (const float*)d_in[8];
  const float* dec_b    = (const float*)d_in[9];
  const float* lin_W    = (const float*)d_in[10];
  const float* lin_b    = (const float*)d_in[11];
  float* out = (float*)d_out;

  const int WN = 4 * Hdim * Hdim;   // 1024*256 elems per weight matrix
  const int HN = Bdim * Hdim;

  size_t off = 0;
  auto alloc = [&](size_t bytes) {
    void* p = (char*)d_ws + off;
    off = (off + bytes + 255) & ~(size_t)255;
    return p;
  };
  unsigned short* wEncWhh0 = (unsigned short*)alloc(WN * 2);
  unsigned short* wEncWhh1 = (unsigned short*)alloc(WN * 2);
  unsigned short* wEncWih1 = (unsigned short*)alloc(WN * 2);
  unsigned short* wDecWhh0 = (unsigned short*)alloc(WN * 2);
  unsigned short* wDecWhh1 = (unsigned short*)alloc(WN * 2);
  unsigned short* wDecWih1 = (unsigned short*)alloc(WN * 2);
  unsigned short* h0buf[2] = {(unsigned short*)alloc(HN * 2), (unsigned short*)alloc(HN * 2)};
  unsigned short* h1buf[2] = {(unsigned short*)alloc(HN * 2), (unsigned short*)alloc(HN * 2)};
  float* c0 = (float*)alloc(HN * 4);
  float* c1 = (float*)alloc(HN * 4);
  float* dec_in = (float*)alloc(Bdim * 3 * 4);

  // 1) bf16 weight packing (weights then stay hot in 192MB L2 for all 192 steps)
  dim3 cb(256), cg((WN + 255) / 256);
  hipLaunchKernelGGL(f2bf_kernel, cg, cb, 0, stream, enc_Whh,            wEncWhh0, WN);
  hipLaunchKernelGGL(f2bf_kernel, cg, cb, 0, stream, enc_Whh + WN,       wEncWhh1, WN);
  hipLaunchKernelGGL(f2bf_kernel, cg, cb, 0, stream, enc_Wih1,           wEncWih1, WN);
  hipLaunchKernelGGL(f2bf_kernel, cg, cb, 0, stream, dec_Whh,            wDecWhh0, WN);
  hipLaunchKernelGGL(f2bf_kernel, cg, cb, 0, stream, dec_Whh + WN,       wDecWhh1, WN);
  hipLaunchKernelGGL(f2bf_kernel, cg, cb, 0, stream, dec_Wih1,           wDecWih1, WN);

  // 2) init state (deterministic every call)
  dim3 zg((HN + 255) / 256);
  hipLaunchKernelGGL(zero_u16_kernel, zg, cb, 0, stream, h0buf[0], HN);
  hipLaunchKernelGGL(zero_u16_kernel, zg, cb, 0, stream, h1buf[0], HN);
  hipLaunchKernelGGL(zero_f32_kernel, zg, cb, 0, stream, c0, HN);
  hipLaunchKernelGGL(zero_f32_kernel, zg, cb, 0, stream, c1, HN);
  hipLaunchKernelGGL(decin_init_kernel, dim3(12), cb, 0, stream, x, dec_in);

  // 1024 waves (64 batch-tiles x 16 hidden-tiles), 4 waves/block
  dim3 gb(128), gg(256);

  // 3) encoder: 128 steps, layer0 then layer1 per step (h ping-pong, c in place)
  for (int t = 0; t < TSRC; ++t) {
    int p = t & 1, q = (t + 1) & 1;
    hipLaunchKernelGGL(lstm_cell_kernel, gg, gb, 0, stream,
                       h0buf[p], wEncWhh0, (const unsigned short*)nullptr,
                       (const unsigned short*)nullptr,
                       x, TSRC * 3, t * 3, enc_Wih0, enc_b, c0, h0buf[q]);
    hipLaunchKernelGGL(lstm_cell_kernel, gg, gb, 0, stream,
                       h1buf[p], wEncWhh1, h0buf[q], wEncWih1,
                       (const float*)nullptr, 0, 0, (const float*)nullptr,
                       enc_b + 4 * Hdim, c1, h1buf[q]);
  }

  // 4) decoder: 64 steps (final encoder state sits in buffer index 0)
  for (int s = 0; s < TTGT; ++s) {
    int p = s & 1, q = (s + 1) & 1;
    hipLaunchKernelGGL(lstm_cell_kernel, gg, gb, 0, stream,
                       h0buf[p], wDecWhh0, (const unsigned short*)nullptr,
                       (const unsigned short*)nullptr,
                       dec_in, 3, 0, dec_Wih0, dec_b, c0, h0buf[q]);
    hipLaunchKernelGGL(lstm_cell_kernel, gg, gb, 0, stream,
                       h1buf[p], wDecWhh1, h0buf[q], wDecWih1,
                       (const float*)nullptr, 0, 0, (const float*)nullptr,
                       dec_b + 4 * Hdim, c1, h1buf[q]);
    hipLaunchKernelGGL(dec_out_kernel, gg, gb, 0, stream,
                       h1buf[q], lin_W, lin_b, dec_in, out, s);
  }
}
